// PatchedInternLM2Attention_4466765988389
// MI455X (gfx1250) — compile-verified, matrix operand order checked by reading, not run on hardware
//
#include <hip/hip_runtime.h>
#include <hip/hip_bf16.h>
#include <stdint.h>

// Problem constants (from reference)
#define BB    4
#define SS    1024
#define HH    4096
#define NKV   8
#define GG    4
#define NQ    32          // NKV * GG
#define DD    128
#define TT    (BB * SS)   // 4096
#define QKVW  6144        // NKV * (GG+2) * DD

typedef __attribute__((ext_vector_type(16))) _Float16 v16h;
typedef __attribute__((ext_vector_type(8)))  _Float16 v8h;
typedef __attribute__((ext_vector_type(8)))  float    v8f;

// ---------------------------------------------------------------------------
// Async global->LDS copy (gfx1250 GLOBAL_LOAD_ASYNC_TO_LDS_B128, ASYNCcnt),
// with a synchronous fallback if the builtin is not declared by the toolchain.
// Probe-confirmed: param 1 is v4i32 in addrspace(1) ("__device__"), so the
// builtin is typed over 16-byte int vectors.
// ---------------------------------------------------------------------------
#if defined(__AMDGCN__) && __has_builtin(__builtin_amdgcn_global_load_async_to_lds_b128)
#define HAVE_ASYNC_LDS 1
#else
#define HAVE_ASYNC_LDS 0
#endif

#if defined(__AMDGCN__) && __has_builtin(__builtin_amdgcn_s_wait_asynccnt)
#define HAVE_WAIT_ASYNC 1
#else
#define HAVE_WAIT_ASYNC 0
#endif

typedef int v4i32 __attribute__((vector_size(16)));
typedef __attribute__((address_space(1))) v4i32 as1_v4i32;
typedef __attribute__((address_space(3))) v4i32 as3_v4i32;

__device__ __forceinline__ void copy16_async(const _Float16* g, _Float16* l) {
#if HAVE_ASYNC_LDS
  __builtin_amdgcn_global_load_async_to_lds_b128(
      (as1_v4i32*)(uintptr_t)g,
      (as3_v4i32*)(unsigned)(uintptr_t)l, 0, 0);
#else
  *reinterpret_cast<v8h*>(l) = *reinterpret_cast<const v8h*>(g);
#endif
}

template <int N>
__device__ __forceinline__ void wait_async() {
#if HAVE_ASYNC_LDS && HAVE_WAIT_ASYNC
  __builtin_amdgcn_s_wait_asynccnt(N);
#endif
}

// ---------------------------------------------------------------------------
// Fragment loader for 16-bit A/B operands of v_wmma_f32_16x16x32_f16.
// Tile is row-major [rows][ldk] f16 in LDS. Per ISA 16-bit A 16x32 layout:
//   lane L holds row (L&15); element e maps to K = e + (e>=8?8:0) + (L>>4)*8
// B (K x N) mirrors with lane = column reading a [N][K] row-major tile.
// ---------------------------------------------------------------------------
__device__ __forceinline__ v16h load_frag(const _Float16* __restrict__ base,
                                          int row_base, int koff, int ldk) {
  const int lane = threadIdx.x & 31;
  const _Float16* p = base + (row_base + (lane & 15)) * ldk + koff + ((lane >> 4) << 3);
  v8h lo = *reinterpret_cast<const v8h*>(p);
  v8h hi = *reinterpret_cast<const v8h*>(p + 16);
  v16h r;
#pragma unroll
  for (int i = 0; i < 8; ++i) { r[i] = lo[i]; r[i + 8] = hi[i]; }
  return r;
}

__device__ __forceinline__ v8f wmma_f16(v16h a, v16h b, v8f c) {
  return __builtin_amdgcn_wmma_f32_16x16x32_f16(false, a, false, b,
                                                (short)0, c, false, false);
}

// Staging helpers: f16 source -> async b128 pair; f32 source -> convert+store.
__device__ __forceinline__ void stage_row(const _Float16* g, _Float16* l) {
  copy16_async(g, l);
  copy16_async(g + 8, l + 8);
}
__device__ __forceinline__ void stage_row(const float* g, _Float16* l) {
  _Float16 t[16];
#pragma unroll
  for (int i = 0; i < 16; ++i) t[i] = (_Float16)g[i];
  *reinterpret_cast<v8h*>(l)     = *reinterpret_cast<v8h*>(&t[0]);
  *reinterpret_cast<v8h*>(l + 8) = *reinterpret_cast<v8h*>(&t[8]);
}

// ---------------------------------------------------------------------------
// Elementwise fp32 -> f16 conversion (one-time, for weights / activations).
// ---------------------------------------------------------------------------
__global__ __launch_bounds__(256)
void cvt_f32_f16(const float* __restrict__ in, _Float16* __restrict__ out, int n) {
  const int i = (blockIdx.x * blockDim.x + threadIdx.x) * 8;
  if (i >= n) return;
  float4 a = *reinterpret_cast<const float4*>(&in[i]);
  float4 b = *reinterpret_cast<const float4*>(&in[i + 4]);
  _Float16 t[8] = {(_Float16)a.x, (_Float16)a.y, (_Float16)a.z, (_Float16)a.w,
                   (_Float16)b.x, (_Float16)b.y, (_Float16)b.z, (_Float16)b.w};
  *reinterpret_cast<v8h*>(&out[i]) = *reinterpret_cast<v8h*>(&t[0]);
}

// ---------------------------------------------------------------------------
// Tiled WMMA GEMM:  C[M,N] = A[M,K] * W[N,K]^T
// Block: 256 threads (8 waves), tile 128(M) x 128(N), K-step 32, double-
// buffered LDS with async global->LDS staging (when both operands are f16).
// Waves: 4 along M (32 rows) x 2 along N (64 cols); 2x4 accumulators each.
// ---------------------------------------------------------------------------
template <typename TA, typename TB, typename TO>
__global__ __launch_bounds__(256)
void gemm_wmma(const TA* __restrict__ A, const TB* __restrict__ W,
               TO* __restrict__ C, int M, int N, int K) {
  __shared__ _Float16 sA[2][128 * 32];
  __shared__ _Float16 sB[2][128 * 32];

  const int tid  = threadIdx.x;
  const int wave = tid >> 5;
  const int lane = tid & 31;
  const int wy   = wave & 3;   // M sub-block (32 rows)
  const int wx   = wave >> 2;  // N sub-block (64 cols)
  const int bm   = blockIdx.y * 128;
  const int bn   = blockIdx.x * 128;

  // async ops issued per thread per stage (2 for each f16 operand)
  constexpr int OPS = (sizeof(TA) == 2 ? 2 : 0) + (sizeof(TB) == 2 ? 2 : 0);

  v8f acc[2][4];
#pragma unroll
  for (int ms = 0; ms < 2; ++ms)
#pragma unroll
    for (int ns = 0; ns < 4; ++ns) acc[ms][ns] = 0.0f;

  const int trow = tid >> 1;           // 0..127
  const int tcol = (tid & 1) << 4;     // 0 or 16

  auto stage = [&](int kt, int buf) {
    stage_row(A + (size_t)(bm + trow) * K + kt * 32 + tcol, &sA[buf][trow * 32 + tcol]);
    stage_row(W + (size_t)(bn + trow) * K + kt * 32 + tcol, &sB[buf][trow * 32 + tcol]);
  };

  const int NKT = K / 32;
  stage(0, 0);
  for (int kt = 0; kt < NKT; ++kt) {
    const int cur = kt & 1;
    if (kt + 1 < NKT) { stage(kt + 1, cur ^ 1); wait_async<OPS>(); }
    else              { wait_async<0>(); }
    __syncthreads();

    v16h af[2], bf[4];
#pragma unroll
    for (int ms = 0; ms < 2; ++ms) af[ms] = load_frag(sA[cur], wy * 32 + ms * 16, 0, 32);
#pragma unroll
    for (int ns = 0; ns < 4; ++ns) bf[ns] = load_frag(sB[cur], wx * 64 + ns * 16, 0, 32);
#pragma unroll
    for (int ms = 0; ms < 2; ++ms)
#pragma unroll
      for (int ns = 0; ns < 4; ++ns)
        acc[ms][ns] = wmma_f16(af[ms], bf[ns], acc[ms][ns]);

    __syncthreads();
  }

  // Epilogue: C/D layout => (m, n) = (i + 8*(lane>=16), lane&15) within tile
  const int cn    = lane & 15;
  const int rhalf = (lane >> 4) << 3;
#pragma unroll
  for (int ms = 0; ms < 2; ++ms) {
#pragma unroll
    for (int ns = 0; ns < 4; ++ns) {
      const int gm = bm + wy * 32 + ms * 16 + rhalf;
      const int gn = bn + wx * 64 + ns * 16 + cn;
#pragma unroll
      for (int i = 0; i < 8; ++i)
        C[(size_t)(gm + i) * N + gn] = (TO)acc[ms][ns][i];
    }
  }
}

// ---------------------------------------------------------------------------
// In-place RoPE on the q (slots 0..G-1) and k (slot G) parts of the f16 qkv
// buffer laid out [T][NKV][(G+2)][D].  One thread per (t, head, pair).
// ---------------------------------------------------------------------------
__global__ __launch_bounds__(256)
void rope_kernel(_Float16* __restrict__ qkv, const int* __restrict__ pos_ids) {
  const int HEADS = NQ + NKV; // 40 roped heads
  const int total = TT * HEADS * (DD / 2);
  int idx = blockIdx.x * blockDim.x + threadIdx.x;
  if (idx >= total) return;

  const int d = idx & 63;
  const int h = (idx >> 6) % HEADS;
  const int t = idx / (HEADS * 64);

  int col;
  if (h < NQ) { col = (h >> 2) * 768 + (h & 3) * 128; }   // q slot
  else        { col = (h - NQ) * 768 + 512; }             // k slot

  _Float16* p = qkv + (size_t)t * QKVW + col;
  const float x1 = (float)p[d];
  const float x2 = (float)p[d + 64];
  const float pos = (float)pos_ids[t];
  const float inv = __powf(1.0e6f, -(float)(2 * d) * (1.0f / 128.0f));
  float s, c;
  __sincosf(pos * inv, &s, &c);
  p[d]      = (_Float16)(x1 * c - x2 * s);
  p[d + 64] = (_Float16)(x2 * c + x1 * s);
}

// ---------------------------------------------------------------------------
// Causal GQA flash attention.  One block = one (query-block of 32, q-head,
// batch); 64 threads = 2 waves, each wave owns 16 query rows.
// Q/K tiles staged with async global->LDS; V staged transposed (sync path);
// QK^T and P*V through WMMA; online softmax state in LDS.
// ---------------------------------------------------------------------------
__global__ __launch_bounds__(64)
void attn_kernel(const _Float16* __restrict__ qkv, _Float16* __restrict__ o) {
  __shared__ _Float16 sQ[32 * 128];
  __shared__ _Float16 sK[64 * 128];
  __shared__ _Float16 sVt[128 * 64];   // transposed: [d][key]
  __shared__ float    sS[32 * 64];
  __shared__ _Float16 sP[32 * 64];
  __shared__ float    sM[32], sL[32], sAl[32];

  const int qb  = blockIdx.x;       // query block (32 rows)
  const int h   = blockIdx.y;       // q head 0..31
  const int b   = blockIdx.z;
  const int kvh = h >> 2;
  const int g   = h & 3;
  const int tid  = threadIdx.x;
  const int wave = tid >> 5;        // 0..1
  const int lane = tid & 31;
  const int qr0  = qb * 32;
  const size_t tbase = (size_t)b * SS;

  const size_t qcol = (size_t)kvh * 768 + g * 128;
  const size_t kcol = (size_t)kvh * 768 + 512;
  const size_t vcol = kcol + 128;

  // Load Q tile (32 x 128) -- async; completion covered by first wait below
  for (int i = tid; i < 32 * 16; i += 64) {
    const int r = i >> 4, c8 = (i & 15) << 3;
    copy16_async(&qkv[(tbase + qr0 + r) * QKVW + qcol + c8], &sQ[r * 128 + c8]);
  }
  if (tid < 32) { sM[tid] = -3.0e38f; sL[tid] = 0.0f; }

  v8f acc[8];
#pragma unroll
  for (int td = 0; td < 8; ++td) acc[td] = 0.0f;

  const int rhalf = (lane >> 4) << 3;
  const int cn    = lane & 15;
  const int nkb   = qr0 / 64 + 1;
  const float scale = 0.08838834764831845f;  // 1/sqrt(128)

  for (int jb = 0; jb < nkb; ++jb) {
    __syncthreads();  // previous iteration fully consumed sK/sVt/sP

    // K tile (64 x 128): async global->LDS
    for (int i = tid; i < 64 * 16; i += 64) {
      const int r = i >> 4, c8 = (i & 15) << 3;
      copy16_async(&qkv[(tbase + jb * 64 + r) * QKVW + kcol + c8], &sK[r * 128 + c8]);
    }
    // V tile transposed into sVt[d][key] (needs shuffle: sync path)
    for (int i = tid; i < 64 * 16; i += 64) {
      const int r = i >> 4, c8 = (i & 15) << 3;
      v8h vv = *reinterpret_cast<const v8h*>(&qkv[(tbase + jb * 64 + r) * QKVW + vcol + c8]);
#pragma unroll
      for (int e = 0; e < 8; ++e) sVt[(c8 + e) * 64 + r] = vv[e];
    }
    wait_async<0>();
    __syncthreads();

    // S = scale * Q K^T  (wave's 16 rows x 64 keys), causal-masked, to LDS
#pragma unroll
    for (int ns = 0; ns < 4; ++ns) {
      v8f s = 0.0f;
#pragma unroll
      for (int kc = 0; kc < 4; ++kc) {
        v16h a  = load_frag(sQ, wave * 16, kc * 32, 128);
        v16h bb = load_frag(sK, ns * 16, kc * 32, 128);
        s = wmma_f16(a, bb, s);
      }
      const int n_g = jb * 64 + ns * 16 + cn;
      const int m_b = qr0 + wave * 16 + rhalf;
#pragma unroll
      for (int i = 0; i < 8; ++i) {
        float v = s[i] * scale;
        if (n_g > m_b + i) v = -1.0e30f;
        sS[(wave * 16 + rhalf + i) * 64 + ns * 16 + cn] = v;
      }
    }
    __syncthreads();

    // Online softmax, one thread per query row
    if (tid < 32) {
      const int m = tid;
      float mx = -3.0e38f;
      for (int j = 0; j < 64; ++j) mx = fmaxf(mx, sS[m * 64 + j]);
      const float newm  = fmaxf(sM[m], mx);
      const float alpha = __expf(sM[m] - newm);
      float sum = 0.0f;
      for (int j = 0; j < 64; ++j) {
        const float p = __expf(sS[m * 64 + j] - newm);
        sum += p;
        sP[m * 64 + j] = (_Float16)p;
      }
      sL[m]  = sL[m] * alpha + sum;
      sM[m]  = newm;
      sAl[m] = alpha;
    }
    __syncthreads();

    // O = alpha * O + P V
    float arow[8];
#pragma unroll
    for (int i = 0; i < 8; ++i) arow[i] = sAl[wave * 16 + rhalf + i];
    v16h pf[2];
#pragma unroll
    for (int kc = 0; kc < 2; ++kc) pf[kc] = load_frag(sP, wave * 16, kc * 32, 64);
#pragma unroll
    for (int td = 0; td < 8; ++td) {
#pragma unroll
      for (int i = 0; i < 8; ++i) acc[td][i] *= arow[i];
#pragma unroll
      for (int kc = 0; kc < 2; ++kc) {
        v16h bb = load_frag(sVt, td * 16, kc * 32, 64);
        acc[td] = wmma_f16(pf[kc], bb, acc[td]);
      }
    }
  }

  // Normalize and store (T, NQ*D) f16
  float rl[8];
#pragma unroll
  for (int i = 0; i < 8; ++i) rl[i] = 1.0f / sL[wave * 16 + rhalf + i];
#pragma unroll
  for (int td = 0; td < 8; ++td) {
#pragma unroll
    for (int i = 0; i < 8; ++i) {
      const size_t row = tbase + qr0 + wave * 16 + rhalf + i;
      o[row * (NQ * DD) + h * DD + td * 16 + cn] = (_Float16)(acc[td][i] * rl[i]);
    }
  }
}

// ---------------------------------------------------------------------------
// Host launcher
// ---------------------------------------------------------------------------
extern "C" void kernel_launch(void* const* d_in, const int* in_sizes, int n_in,
                              void* d_out, int out_size, void* d_ws, size_t ws_size,
                              hipStream_t stream) {
  const float* hidden  = (const float*)d_in[0];  // (1, T, H)
  const float* wqkv    = (const float*)d_in[1];  // (6144, 4096)
  const float* wo      = (const float*)d_in[2];  // (4096, 4096)
  // d_in[3], d_in[4]: k/v caches — scatter/gather round-trip is identity, unused
  // d_in[5]: block_offsets — unused for the returned output
  const int* pos_ids   = (const int*)d_in[6];    // (T,)
  float* out           = (float*)d_out;          // (1, T, H) fp32

  const size_t qkv_b  = (size_t)TT * QKVW * 2;   // f16 qkv
  const size_t o_b    = (size_t)TT * HH * 2;     // f16 attn out
  const size_t h16_b  = (size_t)TT * HH * 2;     // f16 hidden
  const size_t wq16_b = (size_t)QKVW * HH * 2;   // f16 wqkv
  const size_t wo16_b = (size_t)HH * HH * 2;     // f16 wo

  char* p = (char*)d_ws;
  _Float16* qkv = (_Float16*)p;            p += qkv_b;
  _Float16* o   = (_Float16*)p;            p += o_b;

  const bool full = ws_size >= qkv_b + o_b + h16_b + wq16_b + wo16_b;

  if (full) {
    _Float16* h16  = (_Float16*)p;         p += h16_b;
    _Float16* wq16 = (_Float16*)p;         p += wq16_b;
    _Float16* wo16 = (_Float16*)p;

    // 0) one-time fp32 -> f16 conversions
    cvt_f32_f16<<<(TT * HH / 8 + 255) / 256, 256, 0, stream>>>(hidden, h16, TT * HH);
    cvt_f32_f16<<<(QKVW * HH / 8 + 255) / 256, 256, 0, stream>>>(wqkv, wq16, QKVW * HH);
    cvt_f32_f16<<<(HH * HH / 8 + 255) / 256, 256, 0, stream>>>(wo, wo16, HH * HH);

    // 1) qkv = hidden @ wqkv^T   (all-f16 operands -> async LDS staging)
    {
      dim3 grid(QKVW / 128, TT / 128);
      gemm_wmma<_Float16, _Float16, _Float16><<<grid, 256, 0, stream>>>(
          h16, wq16, qkv, TT, QKVW, HH);
    }
    // 2) RoPE in-place
    {
      const int total = TT * (NQ + NKV) * (DD / 2);
      rope_kernel<<<(total + 255) / 256, 256, 0, stream>>>(qkv, pos_ids);
    }
    // 3) Flash attention -> o (f16)
    {
      dim3 grid(SS / 32, NQ, BB);
      attn_kernel<<<grid, 64, 0, stream>>>(qkv, o);
    }
    // 4) out = o @ wo^T
    {
      dim3 grid(HH / 128, TT / 128);
      gemm_wmma<_Float16, _Float16, float><<<grid, 256, 0, stream>>>(
          o, wo16, out, TT, HH, HH);
    }
  } else {
    // Fallback: convert-on-the-fly staging (fp32 operands), no extra ws
    {
      dim3 grid(QKVW / 128, TT / 128);
      gemm_wmma<float, float, _Float16><<<grid, 256, 0, stream>>>(
          hidden, wqkv, qkv, TT, QKVW, HH);
    }
    {
      const int total = TT * (NQ + NKV) * (DD / 2);
      rope_kernel<<<(total + 255) / 256, 256, 0, stream>>>(qkv, pos_ids);
    }
    {
      dim3 grid(SS / 32, NQ, BB);
      attn_kernel<<<grid, 64, 0, stream>>>(qkv, o);
    }
    {
      dim3 grid(HH / 128, TT / 128);
      gemm_wmma<_Float16, float, float><<<grid, 256, 0, stream>>>(
          o, wo, out, TT, HH, HH);
    }
  }
}
